// UberCRSN_24902220382607
// MI455X (gfx1250) — matrix operand order repaired and checked by reading, not compiled, not run on hardware
//
#include <hip/hip_runtime.h>
#include <math.h>

#define BB 64
#define SS 128
#define DD 512
#define MM (BB * SS)            // 8192 rows
#define MD (MM * DD)            // 4194304
#define DEPTH_N 16
#define EPSV 1e-6f
#define THRESHV 0.9999f
#define STK 32

typedef __attribute__((ext_vector_type(16))) __bf16 v16bf;
typedef __attribute__((ext_vector_type(8)))  float  v8f;
typedef __attribute__((ext_vector_type(16))) unsigned short v16us;
typedef __attribute__((ext_vector_type(8)))  unsigned short v8us;

__device__ __forceinline__ unsigned short f2bf(float f) {
  unsigned int x = __builtin_bit_cast(unsigned int, f);
  x += 0x7FFFu + ((x >> 16) & 1u);   // round-to-nearest-even
  return (unsigned short)(x >> 16);
}

__device__ __forceinline__ v8f wmma_bf16(v16bf a, v16bf b, v8f c) {
  return __builtin_amdgcn_wmma_f32_16x16x32_bf16(false, a, false, b, (short)0, c,
                                                 false, false);
}

// A-fragment, 16x32 bf16, row-major A with leading dim lda (elements).
// Lanes 0-15: M=lane, K = 0..15 ; lanes 16-31: M=lane-16, K = 16..31.
__device__ __forceinline__ v16bf load_a_frag(const unsigned short* p0, int lda, int lane) {
  int m = lane & 15;
  int koff = (lane >> 4) << 4;
  const unsigned short* p = p0 + m * lda + koff;
  v8us lo = *reinterpret_cast<const v8us*>(p);
  v8us hi = *reinterpret_cast<const v8us*>(p + 8);
  v16us u;
#pragma unroll
  for (int i = 0; i < 8; ++i) { u[i] = lo[i]; u[i + 8] = hi[i]; }
  return __builtin_bit_cast(v16bf, u);
}

// B-fragment for C = A @ Bm^T : Bm row-major (N x K). B[k][n] = Bm[n][k].
// Lane: n = lane&15, k-half = (lane>>4)*16 -> 32 contiguous bytes of row n.
__device__ __forceinline__ v16bf load_b_frag_nt(const unsigned short* p0, int ldb, int lane) {
  int n = lane & 15;
  int kh = (lane >> 4) << 4;
  const unsigned short* p = p0 + n * ldb + kh;
  v8us lo = *reinterpret_cast<const v8us*>(p);
  v8us hi = *reinterpret_cast<const v8us*>(p + 8);
  v16us u;
#pragma unroll
  for (int i = 0; i < 8; ++i) { u[i] = lo[i]; u[i + 8] = hi[i]; }
  return __builtin_bit_cast(v16bf, u);
}

// B-fragment, K-major B (K x N) with leading dim ldb: strided per-lane gather.
__device__ __forceinline__ v16bf load_b_frag_nn(const unsigned short* p0, int ldb, int lane) {
  int n = lane & 15;
  int kh = (lane >> 4) << 4;
  const unsigned short* p = p0 + kh * ldb + n;
  v16us u;
#pragma unroll
  for (int j = 0; j < 16; ++j) u[j] = p[j * ldb];
  return __builtin_bit_cast(v16bf, u);
}

// ================= complex GEMM (Cr+iCi) = (Ar+iAi) @ (Wr+iWi)^T =================
// Fixed shape: M=MM, N=K=DD. Register-blocked: each wave owns TWO 16x16 N-tiles
// (n0 and n0+128) -> 8 WMMAs per A-fragment pair. Grid: (MM/16)*(DD/256) blocks
// of 256 threads; 8 waves of a block share one M-tile (A reuse through L1/L2).

#define CG_GRID ((MM / 16) * (DD / 256))   // 512 * 2 = 1024

__device__ __forceinline__ void cgemm_core(const unsigned short* __restrict__ Ar,
                                           const unsigned short* __restrict__ Ai,
                                           const unsigned short* __restrict__ Br,
                                           const unsigned short* __restrict__ Bi,
                                           int lane, size_t aoff, size_t b0off, size_t b1off,
                                           v8f acc[8]) {
#pragma unroll 2
  for (int kk = 0; kk < DD; kk += 32) {
    v16bf ar  = load_a_frag(Ar + aoff + kk, DD, lane);
    v16bf ai  = load_a_frag(Ai + aoff + kk, DD, lane);
    v16bf br0 = load_b_frag_nt(Br + b0off + kk, DD, lane);
    v16bf bi0 = load_b_frag_nt(Bi + b0off + kk, DD, lane);
    v16bf br1 = load_b_frag_nt(Br + b1off + kk, DD, lane);
    v16bf bi1 = load_b_frag_nt(Bi + b1off + kk, DD, lane);
    acc[0] = wmma_bf16(ar, br0, acc[0]);   // RR0
    acc[1] = wmma_bf16(ai, bi0, acc[1]);   // II0
    acc[2] = wmma_bf16(ai, br0, acc[2]);   // IR0
    acc[3] = wmma_bf16(ar, bi0, acc[3]);   // RI0
    acc[4] = wmma_bf16(ar, br1, acc[4]);   // RR1
    acc[5] = wmma_bf16(ai, bi1, acc[5]);   // II1
    acc[6] = wmma_bf16(ai, br1, acc[6]);   // IR1
    acc[7] = wmma_bf16(ar, bi1, acc[7]);   // RI1
    __builtin_prefetch(Ar + aoff + kk + 32, 0, 1);   // speculative; OOB harmless
    __builtin_prefetch(Br + b0off + kk + 32, 0, 1);
  }
}

// f32 epilogue (main complex linear -> tr, ti)
__global__ void cgemm_f32(const unsigned short* __restrict__ Ar,
                          const unsigned short* __restrict__ Ai,
                          const unsigned short* __restrict__ Br,
                          const unsigned short* __restrict__ Bi,
                          float* __restrict__ Cr, float* __restrict__ Ci) {
  const int wave = threadIdx.x >> 5;
  const int lane = threadIdx.x & 31;
  const int ngrp = DD >> 8;                                // 2
  const int tileM = blockIdx.x / ngrp;
  const int grp   = blockIdx.x % ngrp;
  const int row0 = tileM << 4;
  const int col0 = (grp * 16 + wave) << 4;
  const int col1 = col0 + 128;
  v8f acc[8] = {};
  cgemm_core(Ar, Ai, Br, Bi, lane,
             (size_t)row0 * DD, (size_t)col0 * DD, (size_t)col1 * DD, acc);
  const int cn = lane & 15;
  const int mo = (lane >> 4) << 3;
#pragma unroll
  for (int v = 0; v < 8; ++v) {
    const int rb = (row0 + mo + v) * DD;
    Cr[rb + col0 + cn] = acc[0][v] - acc[1][v];
    Ci[rb + col0 + cn] = acc[2][v] + acc[3][v];
    Cr[rb + col1 + cn] = acc[4][v] - acc[5][v];
    Ci[rb + col1 + cn] = acc[6][v] + acc[7][v];
  }
}

// bf16 epilogue (q/k/v -> packed attention layouts), row stride ldbf elements
__global__ void cgemm_b16(const unsigned short* __restrict__ Ar,
                          const unsigned short* __restrict__ Ai,
                          const unsigned short* __restrict__ Br,
                          const unsigned short* __restrict__ Bi,
                          unsigned short* __restrict__ Crb,
                          unsigned short* __restrict__ Cib, int ldbf) {
  const int wave = threadIdx.x >> 5;
  const int lane = threadIdx.x & 31;
  const int ngrp = DD >> 8;
  const int tileM = blockIdx.x / ngrp;
  const int grp   = blockIdx.x % ngrp;
  const int row0 = tileM << 4;
  const int col0 = (grp * 16 + wave) << 4;
  const int col1 = col0 + 128;
  v8f acc[8] = {};
  cgemm_core(Ar, Ai, Br, Bi, lane,
             (size_t)row0 * DD, (size_t)col0 * DD, (size_t)col1 * DD, acc);
  const int cn = lane & 15;
  const int mo = (lane >> 4) << 3;
#pragma unroll
  for (int v = 0; v < 8; ++v) {
    const int rb = (row0 + mo + v) * ldbf;
    Crb[rb + col0 + cn] = f2bf(acc[0][v] - acc[1][v]);
    Cib[rb + col0 + cn] = f2bf(acc[2][v] + acc[3][v]);
    Crb[rb + col1 + cn] = f2bf(acc[4][v] - acc[5][v]);
    Cib[rb + col1 + cn] = f2bf(acc[6][v] + acc[7][v]);
  }
}

// scores[b,s,t] = (qf[b,s,:] . kf[b,t,:]) * scale      K = 2*DD
__global__ void scores_gemm(const unsigned short* __restrict__ qf,
                            const unsigned short* __restrict__ kf,
                            float* __restrict__ scores, float scale) {
  const int wave = threadIdx.x >> 5;
  const int lane = threadIdx.x & 31;
  const int tile = blockIdx.x * 8 + wave;     // 64 * 8 * 8 tiles
  const int b = tile >> 6;
  const int r = tile & 63;
  const int tS = r >> 3, tT = r & 7;
  const int K = 2 * DD;
  const unsigned short* A  = qf + (size_t)(b * SS + tS * 16) * K;
  const unsigned short* Bm = kf + (size_t)(b * SS + tT * 16) * K;
  v8f acc = {};
  for (int kk = 0; kk < K; kk += 32) {
    v16bf a  = load_a_frag(A + kk, K, lane);
    v16bf bb = load_b_frag_nt(Bm + kk, K, lane);
    acc = wmma_bf16(a, bb, acc);
    __builtin_prefetch(A + kk + 32, 0, 1);
  }
  const int cn = lane & 15, mo = (lane >> 4) << 3;
#pragma unroll
  for (int v = 0; v < 8; ++v) {
    int s = tS * 16 + mo + v, t = tT * 16 + cn;
    scores[(size_t)(b * SS + s) * SS + t] = acc[v] * scale;
  }
}

// Or/Oi[b,s,d] = sum_t attn[b,s,t] * v{r,i}[b,t,d]
__global__ void attnv_gemm(const unsigned short* __restrict__ attnb,
                           const unsigned short* __restrict__ vrb,
                           const unsigned short* __restrict__ vib,
                           float* __restrict__ Or, float* __restrict__ Oi) {
  const int wave = threadIdx.x >> 5;
  const int lane = threadIdx.x & 31;
  const int tile = blockIdx.x * 8 + wave;     // 64 * 8 * 32 tiles
  const int b = tile >> 8;
  const int r = tile & 255;
  const int tS = r >> 5, tD = r & 31;
  const unsigned short* A  = attnb + (size_t)(b * SS + tS * 16) * SS;
  const unsigned short* Vr = vrb + (size_t)b * SS * DD + tD * 16;
  const unsigned short* Vi = vib + (size_t)b * SS * DD + tD * 16;
  v8f accR = {}, accI = {};
#pragma unroll
  for (int kk = 0; kk < SS; kk += 32) {
    v16bf a  = load_a_frag(A + kk, SS, lane);
    v16bf br = load_b_frag_nn(Vr + (size_t)kk * DD, DD, lane);
    v16bf bi = load_b_frag_nn(Vi + (size_t)kk * DD, DD, lane);
    accR = wmma_bf16(a, br, accR);
    accI = wmma_bf16(a, bi, accI);
  }
  const int cn = lane & 15, mo = (lane >> 4) << 3;
#pragma unroll
  for (int v = 0; v < 8; ++v) {
    int s = tS * 16 + mo + v, d = tD * 16 + cn;
    size_t o = (size_t)(b * SS + s) * DD + d;
    Or[o] = accR[v];
    Oi[o] = accI[v];
  }
}

// ------------------------------ elementwise / reductions ------------------------------

__global__ void fill_f32(float* p, float v, int n) {
  int i = blockIdx.x * blockDim.x + threadIdx.x;
  int stride = gridDim.x * blockDim.x;
  for (; i < n; i += stride) p[i] = v;
}

__global__ void conv_f32_bf16(const float* __restrict__ in, unsigned short* __restrict__ out, int n) {
  int i = blockIdx.x * blockDim.x + threadIdx.x;
  int stride = gridDim.x * blockDim.x;
  for (; i < n; i += stride) out[i] = f2bf(in[i]);
}

__global__ void conv_z(const float* __restrict__ zr, const float* __restrict__ zi,
                       unsigned short* __restrict__ zbr, unsigned short* __restrict__ zbi, int n) {
  int i = blockIdx.x * blockDim.x + threadIdx.x;
  int stride = gridDim.x * blockDim.x;
  for (; i < n; i += stride) { zbr[i] = f2bf(zr[i]); zbi[i] = f2bf(zi[i]); }
}

__global__ void ptrinit(float* p) {
  int gid = blockIdx.x * blockDim.x + threadIdx.x;
  p[gid] = ((gid & 31) == 0) ? 1.f : 0.f;
}

// magnitude layernorm (ddof=1) + phase + focus sigmoid + modrelu -> bf16 activations
__global__ void cellmid(const float* __restrict__ tr, const float* __restrict__ ti,
                        const float* __restrict__ ln_scale, const float* __restrict__ ln_shift,
                        const float* __restrict__ focus_mask, const float* __restrict__ mod_bias,
                        unsigned short* __restrict__ zbr, unsigned short* __restrict__ zbi) {
  const int row = blockIdx.x;
  const int t = threadIdx.x;                       // 256 threads, 2 elems each
  const size_t base = (size_t)row * DD;
  __shared__ float ssum[256], ssq[256];
  __shared__ float smean, srstd;
  float zr0 = tr[base + t],       zi0 = ti[base + t];
  float zr1 = tr[base + t + 256], zi1 = ti[base + t + 256];
  float mag0 = sqrtf(zr0 * zr0 + zi0 * zi0) + EPSV;
  float mag1 = sqrtf(zr1 * zr1 + zi1 * zi1) + EPSV;
  ssum[t] = mag0 + mag1;
  ssq[t]  = mag0 * mag0 + mag1 * mag1;
  __syncthreads();
  for (int off = 128; off > 0; off >>= 1) {
    if (t < off) { ssum[t] += ssum[t + off]; ssq[t] += ssq[t + off]; }
    __syncthreads();
  }
  if (t == 0) {
    float mean = ssum[0] / (float)DD;
    float var  = (ssq[0] - (float)DD * mean * mean) / (float)(DD - 1);
    smean = mean;
    srstd = rsqrtf(var + EPSV);
  }
  __syncthreads();
  const float mean = smean, rstd = srstd;
#pragma unroll
  for (int e = 0; e < 2; ++e) {
    int d = t + e * 256;
    float zr_ = (e == 0) ? zr0 : zr1;
    float zi_ = (e == 0) ? zi0 : zi1;
    float mg  = (e == 0) ? mag0 : mag1;
    float nm = (mg - mean) * rstd * ln_scale[d] + ln_shift[d];
    float cp = zr_ / mg, sp = zi_ / mg;
    float f = 1.f / (1.f + expf(-focus_mask[d]));
    float xr = nm * cp * f, xi = nm * sp * f;
    float norm = sqrtf(xr * xr + xi * xi) + EPSV;
    float s = fmaxf(norm + mod_bias[d], 0.f) / norm;
    zbr[base + d] = f2bf(xr * s);
    zbi[base + d] = f2bf(xi * s);
  }
}

__global__ void softmax_row(float* __restrict__ sc, unsigned short* __restrict__ ab) {
  const int row = blockIdx.x;
  const int t = threadIdx.x;                       // 128 threads
  __shared__ float sm[128];
  __shared__ float red;
  float v = sc[(size_t)row * SS + t];
  sm[t] = v; __syncthreads();
  for (int off = 64; off > 0; off >>= 1) {
    if (t < off) sm[t] = fmaxf(sm[t], sm[t + off]);
    __syncthreads();
  }
  if (t == 0) red = sm[0];
  __syncthreads();
  float e = expf(v - red);
  __syncthreads();
  sm[t] = e; __syncthreads();
  for (int off = 64; off > 0; off >>= 1) {
    if (t < off) sm[t] += sm[t + off];
    __syncthreads();
  }
  if (t == 0) red = sm[0];
  __syncthreads();
  float p = e / red;
  sc[(size_t)row * SS + t] = p;
  ab[(size_t)row * SS + t] = f2bf(p);
}

__global__ void haltctrl(const float* __restrict__ Or, const float* __restrict__ Oi,
                         const float* __restrict__ halt_W, const float* __restrict__ halt_b,
                         const float* __restrict__ ctrl_W, const float* __restrict__ ctrl_b,
                         float* __restrict__ acc, float* __restrict__ wbuf,
                         float* __restrict__ ctrl, int last) {
  const int row = blockIdx.x;
  const int t = threadIdx.x;                       // 256 threads over 2D=1024
  __shared__ float sh[4][256];
  float h = 0.f, c0 = 0.f, c1 = 0.f, c2 = 0.f;
#pragma unroll
  for (int it = 0; it < 4; ++it) {
    int e = t + it * 256;
    float zf = (e < DD) ? Or[(size_t)row * DD + e] : Oi[(size_t)row * DD + (e - DD)];
    h  += zf * halt_W[e];
    c0 += zf * ctrl_W[e * 3 + 0];
    c1 += zf * ctrl_W[e * 3 + 1];
    c2 += zf * ctrl_W[e * 3 + 2];
  }
  sh[0][t] = h; sh[1][t] = c0; sh[2][t] = c1; sh[3][t] = c2;
  __syncthreads();
  for (int off = 128; off > 0; off >>= 1) {
    if (t < off) {
      sh[0][t] += sh[0][t + off]; sh[1][t] += sh[1][t + off];
      sh[2][t] += sh[2][t + off]; sh[3][t] += sh[3][t + off];
    }
    __syncthreads();
  }
  if (t == 0) {
    float hv = 1.f / (1.f + expf(-(sh[0][0] + halt_b[0])));
    float l0 = sh[1][0] + ctrl_b[0];
    float l1 = sh[2][0] + ctrl_b[1];
    float l2 = sh[3][0] + ctrl_b[2];
    float mx = fmaxf(l0, fmaxf(l1, l2));
    float e0 = expf(l0 - mx), e1 = expf(l1 - mx), e2 = expf(l2 - mx);
    float inv = 1.f / (e0 + e1 + e2);
    ctrl[row * 3 + 0] = e0 * inv;
    ctrl[row * 3 + 1] = e1 * inv;
    ctrl[row * 3 + 2] = e2 * inv;
    float old = acc[row];
    float still = (old < THRESHV) ? 1.f : 0.f;
    float wv;
    if (last) wv = (1.f - old) * still;
    else {
      bool wh = (old + hv) >= THRESHV;
      wv = (wh ? (1.f - old) : hv) * still;
    }
    wbuf[row] = wv;
    acc[row] = old + hv * still;
  }
}

__global__ void ctrlmean(const float* __restrict__ ctrl, float* __restrict__ cmean) {
  const int b = blockIdx.x;
  const int t = threadIdx.x;                       // 128 threads
  __shared__ float sm[3][128];
  for (int j = 0; j < 3; ++j) sm[j][t] = ctrl[(size_t)(b * SS + t) * 3 + j];
  __syncthreads();
  for (int off = 64; off > 0; off >>= 1) {
    if (t < off) { sm[0][t] += sm[0][t + off]; sm[1][t] += sm[1][t + off]; sm[2][t] += sm[2][t + off]; }
    __syncthreads();
  }
  if (t == 0) for (int j = 0; j < 3; ++j) cmean[b * 3 + j] = sm[j][0] / (float)SS;
}

__global__ void zpool(const float* __restrict__ Or, const float* __restrict__ Oi,
                      float* __restrict__ zpr, float* __restrict__ zpi) {
  int gid = blockIdx.x * blockDim.x + threadIdx.x;  // BB*DD threads
  int b = gid / DD, d = gid - b * DD;
  float sr = 0.f, si = 0.f;
  for (int s = 0; s < SS; ++s) {
    size_t o = (size_t)(b * SS + s) * DD + d;
    sr += Or[o]; si += Oi[o];
  }
  zpr[gid] = sr / (float)SS;
  zpi[gid] = si / (float)SS;
}

__global__ void stackop(const float* __restrict__ pold, const float* __restrict__ cmean,
                        const float* __restrict__ zpr, const float* __restrict__ zpi,
                        float* __restrict__ memr, float* __restrict__ memi,
                        float* __restrict__ rdr, float* __restrict__ rdi) {
  int gid = blockIdx.x * blockDim.x + threadIdx.x;  // BB*DD threads
  int b = gid / DD, d = gid - b * DD;
  float push = cmean[b * 3 + 0];
  float zr_ = zpr[gid], zi_ = zpi[gid];
  const float* pb = pold + b * STK;
  size_t mb = (size_t)b * STK * DD + d;
  float rr = 0.f, ri = 0.f;
#pragma unroll
  for (int j = 0; j < STK; ++j) {
    float pj = pb[j];
    size_t o = mb + (size_t)j * DD;
    float mr = memr[o], mi = memi[o];
    rr += pj * mr; ri += pj * mi;
    float wm = push * pb[(j + STK - 1) & (STK - 1)];   // ptr rolled by +1
    memr[o] = mr * (1.f - wm) + wm * zr_;
    memi[o] = mi * (1.f - wm) + wm * zi_;
  }
  rdr[gid] = rr; rdi[gid] = ri;
}

__global__ void ptrupd(const float* __restrict__ pold, const float* __restrict__ cmean,
                       float* __restrict__ pnew) {
  int gid = blockIdx.x * blockDim.x + threadIdx.x;  // BB*STK threads
  int b = gid >> 5, j = gid & 31;
  float push = cmean[b * 3 + 0], pop = cmean[b * 3 + 1], noop = cmean[b * 3 + 2];
  const float* pb = pold + b * STK;
  pnew[gid] = push * pb[(j + 31) & 31] + pop * pb[(j + 1) & 31] + noop * pb[j];
}

__global__ void zupdate(const float* __restrict__ Or, const float* __restrict__ Oi,
                        const float* __restrict__ wbuf, const float* __restrict__ cmean,
                        const float* __restrict__ rdr, const float* __restrict__ rdi,
                        float* __restrict__ out, float* __restrict__ zr, float* __restrict__ zi) {
  int idx = blockIdx.x * blockDim.x + threadIdx.x;  // MD threads
  int row = idx / DD, d = idx - row * DD;
  int b = row / SS;
  float nr = Or[idx], ni = Oi[idx];
  float wv = wbuf[row];
  float pop = cmean[b * 3 + 1];
  size_t ob = (size_t)row * (2 * DD) + d;
  out[ob]      += wv * nr;
  out[ob + DD] += wv * ni;
  int rb = b * DD + d;
  zr[idx] = nr + pop * rdr[rb];
  zi[idx] = ni + pop * rdi[rb];
}

// ------------------------------------- host ---------------------------------------

extern "C" void kernel_launch(void* const* d_in, const int* in_sizes, int n_in,
                              void* d_out, int out_size, void* d_ws, size_t ws_size,
                              hipStream_t stream) {
  (void)in_sizes; (void)n_in; (void)out_size; (void)ws_size;
  const float* z_real     = (const float*)d_in[0];
  const float* z_imag     = (const float*)d_in[1];
  const float* Wr         = (const float*)d_in[2];
  const float* Wi         = (const float*)d_in[3];
  const float* ln_scale   = (const float*)d_in[4];
  const float* ln_shift   = (const float*)d_in[5];
  const float* focus_mask = (const float*)d_in[6];
  const float* mod_bias   = (const float*)d_in[7];
  const float* halt_W     = (const float*)d_in[8];
  const float* halt_b     = (const float*)d_in[9];
  const float* ctrl_W     = (const float*)d_in[10];
  const float* ctrl_b     = (const float*)d_in[11];
  const float* qWr        = (const float*)d_in[12];
  const float* qWi        = (const float*)d_in[13];
  const float* kWr        = (const float*)d_in[14];
  const float* kWi        = (const float*)d_in[15];
  const float* vWr        = (const float*)d_in[16];
  const float* vWi        = (const float*)d_in[17];

  char* wptr = (char*)d_ws;
  auto alloc = [&](size_t bytes) -> char* {
    char* p = wptr;
    wptr += (bytes + 255) & ~(size_t)255;
    return p;
  };
  float* zr = (float*)alloc((size_t)MD * 4);
  float* zi = (float*)alloc((size_t)MD * 4);
  float* tr = (float*)alloc((size_t)MD * 4);
  float* ti = (float*)alloc((size_t)MD * 4);
  unsigned short* zbr = (unsigned short*)alloc((size_t)MD * 2);
  unsigned short* zbi = (unsigned short*)alloc((size_t)MD * 2);
  unsigned short* qf  = (unsigned short*)alloc((size_t)MM * 2 * DD * 2);
  unsigned short* kf  = (unsigned short*)alloc((size_t)MM * 2 * DD * 2);
  unsigned short* vrb = (unsigned short*)alloc((size_t)MD * 2);
  unsigned short* vib = (unsigned short*)alloc((size_t)MD * 2);
  float* scores = (float*)alloc((size_t)BB * SS * SS * 4);
  unsigned short* attnb = (unsigned short*)alloc((size_t)BB * SS * SS * 2);
  unsigned short* Wb[8];
  for (int i = 0; i < 8; ++i) Wb[i] = (unsigned short*)alloc((size_t)DD * DD * 2);
  float* acc   = (float*)alloc((size_t)MM * 4);
  float* wbuf  = (float*)alloc((size_t)MM * 4);
  float* ctrl  = (float*)alloc((size_t)MM * 3 * 4);
  float* cmean = (float*)alloc((size_t)BB * 3 * 4);
  float* zpr   = (float*)alloc((size_t)BB * DD * 4);
  float* zpi   = (float*)alloc((size_t)BB * DD * 4);
  float* rdr   = (float*)alloc((size_t)BB * DD * 4);
  float* rdi   = (float*)alloc((size_t)BB * DD * 4);
  float* memr  = (float*)alloc((size_t)BB * STK * DD * 4);
  float* memi  = (float*)alloc((size_t)BB * STK * DD * 4);
  float* ptrA  = (float*)alloc((size_t)BB * STK * 4);
  float* ptrB  = (float*)alloc((size_t)BB * STK * 4);

  float* out = (float*)d_out;

  // ---------- per-call init (harness poisons buffers; re-init everything we read) ----------
  hipMemcpyAsync(zr, z_real, (size_t)MD * 4, hipMemcpyDeviceToDevice, stream);
  hipMemcpyAsync(zi, z_imag, (size_t)MD * 4, hipMemcpyDeviceToDevice, stream);
  fill_f32<<<2048, 256, 0, stream>>>(out, 0.f, MM * 2 * DD);
  fill_f32<<<32, 256, 0, stream>>>(acc, 0.f, MM);
  fill_f32<<<1024, 256, 0, stream>>>(memr, 0.f, BB * STK * DD);
  fill_f32<<<1024, 256, 0, stream>>>(memi, 0.f, BB * STK * DD);
  ptrinit<<<8, 256, 0, stream>>>(ptrA);
  const float* wsrc[8] = {Wr, Wi, qWr, qWi, kWr, kWi, vWr, vWi};
  for (int i = 0; i < 8; ++i)
    conv_f32_bf16<<<256, 256, 0, stream>>>(wsrc[i], Wb[i], DD * DD);

  const float scale = 1.f / sqrtf((float)DD);

  for (int step = 0; step < DEPTH_N; ++step) {
    const float* pold = (step & 1) ? ptrB : ptrA;
    float* pnew       = (step & 1) ? ptrA : ptrB;

    conv_z<<<4096, 256, 0, stream>>>(zr, zi, zbr, zbi, MD);

    // main complex linear: (tr,ti) = z @ W^T
    cgemm_f32<<<CG_GRID, 256, 0, stream>>>(zbr, zbi, Wb[0], Wb[1], tr, ti);

    // magnitude layernorm + focus + modrelu -> bf16 activations for q/k/v
    cellmid<<<MM, 256, 0, stream>>>(tr, ti, ln_scale, ln_shift, focus_mask, mod_bias, zbr, zbi);

    // q / k / v complex linears (bf16 epilogue straight into attention layouts)
    cgemm_b16<<<CG_GRID, 256, 0, stream>>>(zbr, zbi, Wb[2], Wb[3], qf, qf + DD, 2 * DD);
    cgemm_b16<<<CG_GRID, 256, 0, stream>>>(zbr, zbi, Wb[4], Wb[5], kf, kf + DD, 2 * DD);
    cgemm_b16<<<CG_GRID, 256, 0, stream>>>(zbr, zbi, Wb[6], Wb[7], vrb, vib, DD);

    scores_gemm<<<512, 256, 0, stream>>>(qf, kf, scores, scale);
    softmax_row<<<MM, 128, 0, stream>>>(scores, attnb);
    attnv_gemm<<<2048, 256, 0, stream>>>(attnb, vrb, vib, tr, ti);   // or_, oi_ -> tr, ti

    haltctrl<<<MM, 256, 0, stream>>>(tr, ti, halt_W, halt_b, ctrl_W, ctrl_b,
                                     acc, wbuf, ctrl, (step == DEPTH_N - 1) ? 1 : 0);
    ctrlmean<<<BB, 128, 0, stream>>>(ctrl, cmean);
    zpool<<<128, 256, 0, stream>>>(tr, ti, zpr, zpi);
    stackop<<<128, 256, 0, stream>>>(pold, cmean, zpr, zpi, memr, memi, rdr, rdi);
    ptrupd<<<8, 256, 0, stream>>>(pold, cmean, pnew);
    zupdate<<<16384, 256, 0, stream>>>(tr, ti, wbuf, cmean, rdr, rdi, out, zr, zi);
  }
}
